// _SelfAttention_78426102825436
// MI455X (gfx1250) — matrix operand order repaired and checked by reading, not compiled, not run on hardware
//
#include <hip/hip_runtime.h>

#define D_MODEL 1024
#define N_HEADS 16
#define HEAD_DIM 64
#define BATCH 2
#define SEQ 2048
#define NROWS (BATCH * SEQ) // 4096

typedef __attribute__((ext_vector_type(16))) __bf16 v16bf;
typedef __attribute__((ext_vector_type(8)))  __bf16 v8bf;
typedef __attribute__((ext_vector_type(8)))  float  v8f;
typedef __attribute__((ext_vector_type(4)))  int    v4i;

#if defined(__AMDGCN__) && __has_builtin(__builtin_amdgcn_global_load_async_to_lds_b128)
#define USE_ASYNC_LDS 1
#else
#define USE_ASYNC_LDS 0
#endif

__device__ __forceinline__ unsigned short f32_to_bf16(float f) {
    union { float f; unsigned int u; } c; c.f = f;
    unsigned int r = c.u + 0x7FFFu + ((c.u >> 16) & 1u);
    return (unsigned short)(r >> 16);
}

// Stage one 16-byte chunk global -> LDS. Async (ASYNCcnt-tracked, no VGPR
// transit) when the gfx1250 builtin exists, else synchronous copy.
__device__ __forceinline__ void stage16B(const unsigned short* g, unsigned short* l) {
#if USE_ASYNC_LDS
    __builtin_amdgcn_global_load_async_to_lds_b128(
        (__attribute__((address_space(1))) v4i*)g,
        (__attribute__((address_space(3))) v4i*)l, 0, 0);
#else
    *(uint4*)l = *(const uint4*)g;
#endif
}

__device__ __forceinline__ void wait_async_lds() {
#if USE_ASYNC_LDS
    asm volatile("s_wait_asynccnt 0x0" ::: "memory");
#endif
}

// Build a v16bf A/B fragment from two 16-byte LDS/global chunks.
__device__ __forceinline__ v16bf make_frag(const unsigned short* p_lo,
                                           const unsigned short* p_hi) {
    v8bf lo = *(const v8bf*)p_lo;
    v8bf hi = *(const v8bf*)p_hi;
    return __builtin_shufflevector(lo, hi, 0,1,2,3,4,5,6,7,8,9,10,11,12,13,14,15);
}

__device__ __forceinline__ v8f wmma_bf16(v16bf a, v16bf b, v8f c) {
    return __builtin_amdgcn_wmma_f32_16x16x32_bf16(false, a, false, b,
                                                   (short)0, c, false, false);
}

// Lane exchange with immediate XOR pattern (group-of-32 ds_swizzle: no index
// math, unlike __shfl_xor's ds_bpermute path).
template <int XM>
__device__ __forceinline__ float swz_xor(float v) {
    int i = __builtin_amdgcn_ds_swizzle(__float_as_int(v), (XM << 10) | 0x1f);
    return __int_as_float(i);
}
__device__ __forceinline__ float red_max16(float v) {
    v = fmaxf(v, swz_xor<1>(v));
    v = fmaxf(v, swz_xor<2>(v));
    v = fmaxf(v, swz_xor<4>(v));
    v = fmaxf(v, swz_xor<8>(v));
    return v;
}
__device__ __forceinline__ float red_sum16(float v) {
    v += swz_xor<1>(v);
    v += swz_xor<2>(v);
    v += swz_xor<4>(v);
    v += swz_xor<8>(v);
    return v;
}

// ---------------------------------------------------------------------------
// fp32 -> bf16 conversion (2 elements / thread)
// ---------------------------------------------------------------------------
__global__ void k_f32_to_bf16(const float* __restrict__ in,
                              unsigned short* __restrict__ out, int n2) {
    int i = blockIdx.x * blockDim.x + threadIdx.x;
    if (i < n2) {
        float2 v = ((const float2*)in)[i];
        unsigned int p = (unsigned int)f32_to_bf16(v.x) |
                         ((unsigned int)f32_to_bf16(v.y) << 16);
        ((unsigned int*)out)[i] = p;
    }
}

// ---------------------------------------------------------------------------
// C[M,N] = A[M,K] * B[N,K]^T   (bf16 in, f32 accumulate, OutT out)
// Block: 256 threads (8 waves), tile 128x128x32, double-buffered async LDS
// staging. Wave (wm,wn) in 2x4 grid: 64 rows x 32 cols = 4x2 WMMA tiles.
// ---------------------------------------------------------------------------
template <typename OutT>
__global__ __launch_bounds__(256)
void k_gemm_bf16_nt(const unsigned short* __restrict__ A,
                    const unsigned short* __restrict__ B,
                    OutT* __restrict__ C, int M, int N, int K) {
    __shared__ unsigned short As[2][128][40];
    __shared__ unsigned short Bs[2][128][40];

    const int t    = threadIdx.x;
    const int lane = t & 31;
    const int wave = t >> 5;
    const int wm   = wave >> 2;           // 0..1
    const int wn   = wave & 3;            // 0..3
    const int ln16 = lane & 15;
    const int hi16 = (lane >> 4) & 1;     // 0 / 1
    const int m0   = blockIdx.y * 128;
    const int n0   = blockIdx.x * 128;

    v8f acc[4][2];
#pragma unroll
    for (int a = 0; a < 4; a++)
#pragma unroll
        for (int b = 0; b < 2; b++) acc[a][b] = (v8f)0.0f;

    auto stage_tile = [&](int kt, int p) {
        const int k0 = kt * 32;
#pragma unroll
        for (int i = 0; i < 2; i++) {
            int c   = t + 256 * i;        // 0..511
            int row = c >> 2;             // 0..127
            int seg = c & 3;              // 0..3 (8 halves each)
            stage16B(A + (size_t)(m0 + row) * K + k0 + seg * 8, &As[p][row][seg * 8]);
            stage16B(B + (size_t)(n0 + row) * K + k0 + seg * 8, &Bs[p][row][seg * 8]);
        }
    };

    const int nk = K / 32;
    stage_tile(0, 0);

    for (int kt = 0; kt < nk; kt++) {
        const int p = kt & 1;
        wait_async_lds();
        __syncthreads();                  // tile kt resident for all waves
        if (kt + 1 < nk) stage_tile(kt + 1, p ^ 1);   // overlap with compute

        v16bf af[4], bfr[2];
        {
            int off = hi16 * 8;
#pragma unroll
            for (int mt = 0; mt < 4; mt++) {
                int row = wm * 64 + mt * 16 + ln16;
                af[mt] = make_frag(&As[p][row][off], &As[p][row][16 + off]);
            }
        }
        {
            int off = hi16 * 8;
#pragma unroll
            for (int nt = 0; nt < 2; nt++) {
                int col = wn * 32 + nt * 16 + ln16;
                bfr[nt] = make_frag(&Bs[p][col][off], &Bs[p][col][off + 16]);
            }
        }
#pragma unroll
        for (int mt = 0; mt < 4; mt++)
#pragma unroll
            for (int nt = 0; nt < 2; nt++)
                acc[mt][nt] = wmma_bf16(af[mt], bfr[nt], acc[mt][nt]);

        __syncthreads();                  // done reading buffer p
    }

#pragma unroll
    for (int mt = 0; mt < 4; mt++)
#pragma unroll
        for (int nt = 0; nt < 2; nt++) {
            int colg = n0 + wn * 32 + nt * 16 + ln16;
#pragma unroll
            for (int r = 0; r < 8; r++) {
                int   rowg = m0 + wm * 64 + mt * 16 + r + hi16 * 8;
                float v    = acc[mt][nt][r];
                if constexpr (sizeof(OutT) == 2)
                    C[(size_t)rowg * N + colg] = (OutT)f32_to_bf16(v);
                else
                    C[(size_t)rowg * N + colg] = (OutT)v;
            }
        }
}

// ---------------------------------------------------------------------------
// Causal flash attention. qkv: [B*T, 3*D_MODEL] bf16 (q|k|v blocks).
// Grid: (B*H, T/64). Block: 128 threads (4 waves); wave w handles 16 query
// rows. Key tiles of 32; S and P*V both via v_wmma_f32_16x16x32_bf16.
// K tile staged via async LDS path; V transposed manually (overlaps async).
// ---------------------------------------------------------------------------
__global__ __launch_bounds__(128)
void k_flash_attn(const unsigned short* __restrict__ qkv,
                  unsigned short* __restrict__ attn_out) {
    __shared__ unsigned short Ks[32][72];      // [key][d]
    __shared__ unsigned short Vts[64][40];     // [d][key] (transposed)
    __shared__ unsigned short Ps[4][16][40];   // per-wave P scratch [q][key]

    const int bh   = blockIdx.x;
    const int b    = bh / N_HEADS;
    const int h    = bh % N_HEADS;
    const int q0   = blockIdx.y * 64;
    const int t    = threadIdx.x;
    const int lane = t & 31;
    const int wave = t >> 5;
    const int ln16 = lane & 15;
    const int hi16 = (lane >> 4) & 1;

    const size_t rs = 3 * D_MODEL;
    const unsigned short* qbase = qkv + (size_t)b * SEQ * rs + (size_t)h * HEAD_DIM;
    const unsigned short* kbase = qbase + D_MODEL;
    const unsigned short* vbase = qbase + 2 * D_MODEL;

    // Q fragments (A layout), rows q0 + wave*16 + ln16, d split 0-31 / 32-63.
    v16bf qf[2];
    {
        const unsigned short* qp = qbase + (size_t)(q0 + wave * 16 + ln16) * rs;
        int off = hi16 * 8;
        qf[0] = make_frag(qp + off,      qp + 16 + off);
        qf[1] = make_frag(qp + 32 + off, qp + 48 + off);
    }

    v8f oacc[4];
#pragma unroll
    for (int n = 0; n < 4; n++) oacc[n] = (v8f)0.0f;
    float mrow[8], lrow[8];
#pragma unroll
    for (int r = 0; r < 8; r++) { mrow[r] = -1e30f; lrow[r] = 0.0f; }

    const float scale  = 0.125f;               // 1/sqrt(64)
    const int   qhi    = q0 + wave * 16 + 15;  // wave-uniform
    const int   ntiles = (q0 + 64) / 32;

    for (int kt = 0; kt < ntiles; kt++) {
        __syncthreads();
        {   // stage K (async, natural) + V (manual transpose): 32 rows x 4 segs
            int row = t >> 2;
            int seg = t & 3;
            const unsigned short* kp = kbase + (size_t)(kt * 32 + row) * rs + seg * 16;
            stage16B(kp,     &Ks[row][seg * 16]);
            stage16B(kp + 8, &Ks[row][seg * 16 + 8]);
            const unsigned short* vp = vbase + (size_t)(kt * 32 + row) * rs + seg * 16;
            uint4 v1 = *(const uint4*)vp;
            uint4 v2 = *(const uint4*)(vp + 8);
            const unsigned short* e1 = (const unsigned short*)&v1;
            const unsigned short* e2 = (const unsigned short*)&v2;
#pragma unroll
            for (int i = 0; i < 8; i++) Vts[seg * 16 + i][row]     = e1[i];
#pragma unroll
            for (int i = 0; i < 8; i++) Vts[seg * 16 + 8 + i][row] = e2[i];
        }
        wait_async_lds();
        __syncthreads();

        if (kt * 32 <= qhi) {   // wave-uniform skip of fully-masked tiles
            // ---- S = Q K^T  (two 16-key halves, K-dim = 64 = 2 WMMAs each)
            v8f s1 = (v8f)0.0f, s2 = (v8f)0.0f;
            {
                int off = hi16 * 8;
                v16bf kf0 = make_frag(&Ks[ln16][off],          &Ks[ln16][off + 16]);
                v16bf kf1 = make_frag(&Ks[ln16][32 + off],     &Ks[ln16][32 + off + 16]);
                s1 = wmma_bf16(qf[0], kf0, s1);
                s1 = wmma_bf16(qf[1], kf1, s1);
                v16bf kg0 = make_frag(&Ks[16 + ln16][off],      &Ks[16 + ln16][off + 16]);
                v16bf kg1 = make_frag(&Ks[16 + ln16][32 + off], &Ks[16 + ln16][32 + off + 16]);
                s2 = wmma_bf16(qf[0], kg0, s2);
                s2 = wmma_bf16(qf[1], kg1, s2);
            }

            // ---- online softmax (C layout: elem r <-> row r + 8*hi16, col ln16)
            float p1[8], p2[8], alpha[8];
#pragma unroll
            for (int r = 0; r < 8; r++) {
                int   qg  = q0 + wave * 16 + r + hi16 * 8;
                float a   = s1[r] * scale;
                float c2  = s2[r] * scale;
                int   kg1 = kt * 32 + ln16;
                if (kg1 > qg)      a  = -__builtin_inff();
                if (kg1 + 16 > qg) c2 = -__builtin_inff();
                float mt2  = red_max16(fmaxf(a, c2));
                float mnew = fmaxf(mrow[r], mt2);
                alpha[r] = __expf(mrow[r] - mnew);
                mrow[r]  = mnew;
                a  = __expf(a - mnew);
                c2 = __expf(c2 - mnew);
                lrow[r] = lrow[r] * alpha[r] + red_sum16(a + c2);
                p1[r] = a; p2[r] = c2;
            }
#pragma unroll
            for (int n = 0; n < 4; n++)
#pragma unroll
                for (int r = 0; r < 8; r++) oacc[n][r] = oacc[n][r] * alpha[r];

            // ---- repack P (C layout -> A layout) through per-wave LDS
#pragma unroll
            for (int r = 0; r < 8; r++) {
                Ps[wave][r + hi16 * 8][ln16]      = f32_to_bf16(p1[r]);
                Ps[wave][r + hi16 * 8][16 + ln16] = f32_to_bf16(p2[r]);
            }
            int off8 = hi16 * 8;
            v16bf pf = make_frag(&Ps[wave][ln16][off8], &Ps[wave][ln16][16 + off8]);

            // ---- O += P * V  (4 d-tiles of 16)
            int offv = hi16 * 8;
#pragma unroll
            for (int n = 0; n < 4; n++) {
                v16bf vf = make_frag(&Vts[n * 16 + ln16][offv],
                                     &Vts[n * 16 + ln16][offv + 16]);
                oacc[n] = wmma_bf16(pf, vf, oacc[n]);
            }
        }
    }

    // ---- normalize and store (bf16, [B*T, D_MODEL] with head offset)
    float inv[8];
#pragma unroll
    for (int r = 0; r < 8; r++) inv[r] = 1.0f / lrow[r];
#pragma unroll
    for (int n = 0; n < 4; n++) {
        int d = n * 16 + ln16;
#pragma unroll
        for (int r = 0; r < 8; r++) {
            int q = q0 + wave * 16 + r + hi16 * 8;
            attn_out[((size_t)(b * SEQ + q)) * D_MODEL + h * HEAD_DIM + d] =
                f32_to_bf16(oacc[n][r] * inv[r]);
        }
    }
}

// ---------------------------------------------------------------------------
extern "C" void kernel_launch(void* const* d_in, const int* in_sizes, int n_in,
                              void* d_out, int out_size, void* d_ws, size_t ws_size,
                              hipStream_t stream) {
    const float* x    = (const float*)d_in[0];   // [2,2048,1024]
    const float* Wqkv = (const float*)d_in[1];   // [3072,1024]
    const float* Wout = (const float*)d_in[2];   // [1024,1024]
    float* out = (float*)d_out;                  // [2,2048,1024]

    unsigned short* xb    = (unsigned short*)d_ws;
    unsigned short* wqkvb = xb    + (size_t)NROWS * D_MODEL;
    unsigned short* woutb = wqkvb + (size_t)3 * D_MODEL * D_MODEL;
    unsigned short* qkvb  = woutb + (size_t)D_MODEL * D_MODEL;
    unsigned short* attnb = qkvb  + (size_t)NROWS * 3 * D_MODEL;

    const int n_x  = NROWS * D_MODEL;
    const int n_wq = 3 * D_MODEL * D_MODEL;
    const int n_wo = D_MODEL * D_MODEL;
    k_f32_to_bf16<<<(n_x / 2 + 255) / 256, 256, 0, stream>>>(x, xb, n_x / 2);
    k_f32_to_bf16<<<(n_wq / 2 + 255) / 256, 256, 0, stream>>>(Wqkv, wqkvb, n_wq / 2);
    k_f32_to_bf16<<<(n_wo / 2 + 255) / 256, 256, 0, stream>>>(Wout, woutb, n_wo / 2);

    dim3 g1(3 * D_MODEL / 128, NROWS / 128);   // (24, 32)
    k_gemm_bf16_nt<unsigned short><<<g1, 256, 0, stream>>>(
        xb, wqkvb, qkvb, NROWS, 3 * D_MODEL, D_MODEL);

    dim3 g2(BATCH * N_HEADS, SEQ / 64);        // (32, 32)
    k_flash_attn<<<g2, 128, 0, stream>>>(qkvb, attnb);

    dim3 g3(D_MODEL / 128, NROWS / 128);       // (8, 32)
    k_gemm_bf16_nt<float><<<g3, 256, 0, stream>>>(
        attnb, woutb, out, NROWS, D_MODEL, D_MODEL);
}
